// CRF_79156247265721
// MI455X (gfx1250) — matrix operand order repaired and checked by reading, not compiled, not run on hardware
//
#include <hip/hip_runtime.h>
#include <hip/hip_bf16.h>

typedef _Float16 v8h  __attribute__((ext_vector_type(8)));
typedef _Float16 v16h __attribute__((ext_vector_type(16)));
typedef float    v8f  __attribute__((ext_vector_type(8)));

typedef unsigned int uint32x4 __attribute__((ext_vector_type(4)));
typedef int          int32x4  __attribute__((ext_vector_type(4)));
typedef int          int32x8  __attribute__((ext_vector_type(8)));

#define CRF_S 512
#define CRF_B 1024
#define CRF_T 64
#define FWD_WAVES 4   // waves per block; each wave owns a 16-batch tile

#if defined(__has_builtin)
#if __has_builtin(__builtin_amdgcn_tensor_load_to_lds) && __has_builtin(__builtin_amdgcn_s_wait_tensorcnt)
#define USE_TDM 1
#endif
#endif
#ifndef USE_TDM
#define USE_TDM 0
#endif

#if USE_TDM
// Issue a TDM DMA of one contiguous 4KB tile (1024 f32) global -> LDS.
// D# layout per CDNA5 ISA ch.8 (group0: count/lds_addr/global_addr/type,
// group1: data_size=4B, tensor_dim0=tile_dim0=1024, 1-D copy).
__device__ static inline void tdm_load_tile_4kb(const float* gptr, unsigned lds_byte_addr) {
    const unsigned long long ga = (unsigned long long)(uintptr_t)gptr;

    unsigned g0w0 = 1u;                                        // count=1, user desc
    unsigned g0w1 = lds_byte_addr;                             // lds_addr
    unsigned g0w2 = (unsigned)ga;                              // global_addr[31:0]
    unsigned g0w3 = (unsigned)((ga >> 32) & 0x01FFFFFFu)       // global_addr[56:32]
                    | (2u << 30);                              // type=2 ("image")

    unsigned g1w0 = 2u << 16;                                  // data_size=2 -> 4 bytes
    unsigned g1w1 = (1024u & 0xFFFFu) << 16;                   // tensor_dim0[15:0]
    unsigned g1w2 = (1024u >> 16) | (1u << 16);                // tensor_dim0[31:16] | tensor_dim1[15:0]=1
    unsigned g1w3 = (1024u << 16);                             // tensor_dim1[31:16]=0 | tile_dim0=1024
    unsigned g1w4 = 1u;                                        // tile_dim1=1, tile_dim2=0
    unsigned g1w5 = 1024u;                                     // tensor_dim0_stride[31:0]
    unsigned g1w6 = (1024u & 0xFFFFu) << 16;                   // t0s[47:32]=0 | t1s[15:0]
    unsigned g1w7 = 0u;                                        // t1s[47:16]=0

    // force descriptor words into SGPRs
    g0w0 = (unsigned)__builtin_amdgcn_readfirstlane((int)g0w0);
    g0w1 = (unsigned)__builtin_amdgcn_readfirstlane((int)g0w1);
    g0w2 = (unsigned)__builtin_amdgcn_readfirstlane((int)g0w2);
    g0w3 = (unsigned)__builtin_amdgcn_readfirstlane((int)g0w3);
    g1w0 = (unsigned)__builtin_amdgcn_readfirstlane((int)g1w0);
    g1w1 = (unsigned)__builtin_amdgcn_readfirstlane((int)g1w1);
    g1w2 = (unsigned)__builtin_amdgcn_readfirstlane((int)g1w2);
    g1w3 = (unsigned)__builtin_amdgcn_readfirstlane((int)g1w3);
    g1w4 = (unsigned)__builtin_amdgcn_readfirstlane((int)g1w4);
    g1w5 = (unsigned)__builtin_amdgcn_readfirstlane((int)g1w5);
    g1w6 = (unsigned)__builtin_amdgcn_readfirstlane((int)g1w6);
    g1w7 = (unsigned)__builtin_amdgcn_readfirstlane((int)g1w7);

    uint32x4 g0 = { g0w0, g0w1, g0w2, g0w3 };
    int32x8  g1 = { (int)g1w0, (int)g1w1, (int)g1w2, (int)g1w3,
                    (int)g1w4, (int)g1w5, (int)g1w6, (int)g1w7 };
    int32x4  z4 = { 0, 0, 0, 0 };
#if defined(__clang_major__) && (__clang_major__ >= 23)
    int32x8  z8 = { 0, 0, 0, 0, 0, 0, 0, 0 };
    __builtin_amdgcn_tensor_load_to_lds(g0, g1, z4, z4, z8, 0);
#else
    __builtin_amdgcn_tensor_load_to_lds(g0, g1, z4, z4, 0);
#endif
}
#endif // USE_TDM

// ---------------------------------------------------------------------------
// Forward (denominator) kernel: log-space scan turned into f16 WMMA GEMMs.
// new_alpha[b,j] = emit[s,b,j] + rowmax_b + log( sum_i exp(alpha[b,i]-rowmax_b) * exp(trans[i,j]) )
// ---------------------------------------------------------------------------
__global__ __launch_bounds__(FWD_WAVES * 32) void crf_forward_wmma(
    const float* __restrict__ emissions,          // (S,B,T)
    const unsigned char* __restrict__ mask,       // (S,B) bool bytes
    const float* __restrict__ start_transitions,  // (T)
    const float* __restrict__ end_transitions,    // (T)
    const float* __restrict__ transitions,        // (T,T)
    float* __restrict__ den_out)                  // (B)
{
    __shared__ __align__(32) _Float16 Et[CRF_T][CRF_T];            // Et[n][k] = exp(trans[k][n])
    __shared__ __align__(32) float    alpha[FWD_WAVES][16][CRF_T]; // per-wave alpha tile
    __shared__ __align__(32) _Float16 Pt[FWD_WAVES][16][CRF_T];    // per-wave p tile (f16)
    __shared__ float redbuf[FWD_WAVES][32];
    __shared__ float rowmaxS[FWD_WAVES][16];
#if USE_TDM
    __shared__ __align__(16) float EmitBuf[FWD_WAVES][16 * CRF_T]; // TDM staging (4KB/wave)
#endif

    const int tid  = threadIdx.x;
    const int w    = tid >> 5;
    const int lane = tid & 31;
    const int b0   = blockIdx.x * (FWD_WAVES * 16) + w * 16;

    // Build Et = exp(transitions)^T once per block (transitions[i]=trans[k*64+n]).
    for (int i = tid; i < CRF_T * CRF_T; i += FWD_WAVES * 32) {
        const int k = i >> 6, n = i & 63;
        Et[n][k] = (_Float16)__expf(transitions[i]);
    }
    __syncthreads();

    const int l15 = lane & 15;   // A: row M ; B/D: column-in-tile
    const int h   = lane >> 4;   // lane half

    // Preload constant B operands (exp(trans)) into registers for the whole scan.
    v16h Bop[8];
#pragma unroll
    for (int kt = 0; kt < 2; ++kt)
#pragma unroll
        for (int nt = 0; nt < 4; ++nt)
            Bop[kt * 4 + nt] = __builtin_shufflevector(*(const v8h*)&Et[nt * 16 + l15][kt * 32 + 8 * h], *(const v8h*)&Et[nt * 16 + l15][kt * 32 + 16 + 8 * h], 0,1,2,3,4,5,6,7,8,9,10,11,12,13,14,15);

    const int row   = lane >> 1;        // 2 lanes per alpha row
    const int chalf = (lane & 1) * 32;  // column half handled by this lane

    // alpha0 = start_transitions + emissions[0]
    {
        const float* em0 = emissions + (size_t)(b0 + row) * CRF_T;
#pragma unroll 8
        for (int c = 0; c < 32; ++c) {
            const int cc = chalf + c;
            alpha[w][row][cc] = start_transitions[cc] + em0[cc];
        }
    }
    __syncthreads();

#if USE_TDM
    // Kick off DMA of step-1 emissions tile (contiguous 16x64 f32 = 4KB).
    tdm_load_tile_4kb(emissions + (size_t)1 * (CRF_B * CRF_T) + (size_t)b0 * CRF_T,
                      (unsigned)(uintptr_t)(void*)&EmitBuf[w][0]);
#endif

    for (int s = 1; s < CRF_S; ++s) {
        // ---- phase 1: row max of alpha tile ----
        float mx = -1e30f;
#pragma unroll 8
        for (int c = 0; c < 32; ++c) mx = fmaxf(mx, alpha[w][row][chalf + c]);
        redbuf[w][lane] = mx;
        __syncthreads();
        const float rmax = fmaxf(redbuf[w][2 * row], redbuf[w][2 * row + 1]);
        if ((lane & 1) == 0) rowmaxS[w][row] = rmax;

        // ---- phase 2: p = exp(alpha - rowmax) -> f16 tile ----
#pragma unroll 8
        for (int c = 0; c < 32; ++c) {
            const int cc = chalf + c;
            Pt[w][row][cc] = (_Float16)__expf(alpha[w][row][cc] - rmax);
        }
        __syncthreads();

        // ---- phase 3: (16x64 p) @ (64x64 E) via 8 chained WMMAs ----
        const v8h lo0 = *(const v8h*)&Pt[w][l15][8 * h];
        const v8h hi0 = *(const v8h*)&Pt[w][l15][16 + 8 * h];
        const v8h lo1 = *(const v8h*)&Pt[w][l15][32 + 8 * h];
        const v8h hi1 = *(const v8h*)&Pt[w][l15][48 + 8 * h];
        const v16h A0 = __builtin_shufflevector(lo0, hi0, 0,1,2,3,4,5,6,7,8,9,10,11,12,13,14,15);
        const v16h A1 = __builtin_shufflevector(lo1, hi1, 0,1,2,3,4,5,6,7,8,9,10,11,12,13,14,15);

        v8f acc[4];
#pragma unroll
        for (int nt = 0; nt < 4; ++nt) {
            v8f c = {};
            c = __builtin_amdgcn_wmma_f32_16x16x32_f16(false, A0, false, Bop[nt],
                                                       (short)0, c, false, false);
            c = __builtin_amdgcn_wmma_f32_16x16x32_f16(false, A1, false, Bop[4 + nt],
                                                       (short)0, c, false, false);
            acc[nt] = c;
        }

#if USE_TDM
        __builtin_amdgcn_s_wait_tensorcnt((short)0);   // this step's tile is in LDS
        __builtin_amdgcn_fence(__ATOMIC_ACQUIRE, "workgroup");
#endif

        // ---- phase 4: alpha = mask ? emit + rowmax + log(sum) : alpha ----
        const size_t sbase = (size_t)s * (CRF_B * CRF_T);
        (void)sbase;
#pragma unroll
        for (int nt = 0; nt < 4; ++nt) {
            const int n = nt * 16 + l15;
#pragma unroll
            for (int r = 0; r < 8; ++r) {
                const int mm   = r + 8 * h;
                const float oldv = alpha[w][mm][n];
#if USE_TDM
                const float emit = EmitBuf[w][mm * CRF_T + n];
#else
                const float emit = emissions[sbase + (size_t)(b0 + mm) * CRF_T + n];
#endif
                const float nv   = emit + rowmaxS[w][mm] + __logf(acc[nt][r]);
                const unsigned char mk = mask[(size_t)s * CRF_B + (b0 + mm)];
                alpha[w][mm][n] = mk ? nv : oldv;
            }
        }
        __syncthreads();

#if USE_TDM
        if (s + 1 < CRF_S) {
            tdm_load_tile_4kb(emissions + (size_t)(s + 1) * (CRF_B * CRF_T) + (size_t)b0 * CRF_T,
                              (unsigned)(uintptr_t)(void*)&EmitBuf[w][0]);
        }
#endif
    }

    // ---- final: den[b] = logsumexp_t(alpha[b,t] + end_transitions[t]) ----
    float mx = -1e30f;
#pragma unroll 8
    for (int c = 0; c < 32; ++c) {
        const int cc = chalf + c;
        mx = fmaxf(mx, alpha[w][row][cc] + end_transitions[cc]);
    }
    redbuf[w][lane] = mx;
    __syncthreads();
    const float rmax = fmaxf(redbuf[w][2 * row], redbuf[w][2 * row + 1]);
    float sum = 0.f;
#pragma unroll 8
    for (int c = 0; c < 32; ++c) {
        const int cc = chalf + c;
        sum += __expf(alpha[w][row][cc] + end_transitions[cc] - rmax);
    }
    __syncthreads();
    redbuf[w][lane] = sum;
    __syncthreads();
    if ((lane & 1) == 0)
        den_out[b0 + row] = rmax + __logf(redbuf[w][2 * row] + redbuf[w][2 * row + 1]);
}

// ---------------------------------------------------------------------------
// Numerator (gold-path score) kernel: one thread per batch element.
// ---------------------------------------------------------------------------
__global__ __launch_bounds__(256) void crf_numerator(
    const float* __restrict__ emissions,
    const int* __restrict__ tags,            // (S,B) int
    const unsigned char* __restrict__ mask,  // (S,B) bool bytes
    const float* __restrict__ start_transitions,
    const float* __restrict__ end_transitions,
    const float* __restrict__ transitions,
    float* __restrict__ num_out)
{
    const int b = blockIdx.x * blockDim.x + threadIdx.x;
    if (b >= CRF_B) return;

    int prev = tags[b];
    float score = start_transitions[prev];
    float acc = 0.f;
    int cnt = 0;
    for (int s = 0; s < CRF_S - 1; ++s) {
        const int nxt = tags[(size_t)(s + 1) * CRF_B + b];
        const float mk_s  = mask[(size_t)s * CRF_B + b] ? 1.f : 0.f;
        const float mk_s1 = mask[(size_t)(s + 1) * CRF_B + b] ? 1.f : 0.f;
        acc += transitions[prev * CRF_T + nxt] * mk_s1;
        acc += emissions[(size_t)s * (CRF_B * CRF_T) + (size_t)b * CRF_T + prev] * mk_s;
        cnt += (mk_s != 0.f) ? 1 : 0;
        prev = nxt;
    }
    const float mkL = mask[(size_t)(CRF_S - 1) * CRF_B + b] ? 1.f : 0.f;
    cnt += (mkL != 0.f) ? 1 : 0;
    const int last_idx = cnt - 1;
    const int last_tag = tags[(size_t)last_idx * CRF_B + b];
    score += acc + end_transitions[last_tag] +
             emissions[(size_t)(CRF_S - 1) * (CRF_B * CRF_T) + (size_t)b * CRF_T + last_tag] * mkL;
    num_out[b] = score;
}

// ---------------------------------------------------------------------------
// Final reduction: sum_b (num[b] - den[b]) -> scalar
// ---------------------------------------------------------------------------
__global__ __launch_bounds__(256) void crf_reduce(
    const float* __restrict__ num, const float* __restrict__ den,
    float* __restrict__ out)
{
    __shared__ float sh[256];
    const int t = threadIdx.x;
    float s = 0.f;
    for (int i = t; i < CRF_B; i += 256) s += num[i] - den[i];
    sh[t] = s;
    __syncthreads();
    for (int off = 128; off > 0; off >>= 1) {
        if (t < off) sh[t] += sh[t + off];
        __syncthreads();
    }
    if (t == 0) out[0] = sh[0];
}

extern "C" void kernel_launch(void* const* d_in, const int* in_sizes, int n_in,
                              void* d_out, int out_size, void* d_ws, size_t ws_size,
                              hipStream_t stream) {
    (void)in_sizes; (void)n_in; (void)out_size; (void)ws_size;
    const float*         emissions = (const float*)d_in[0];
    const int*           tags      = (const int*)d_in[1];
    const unsigned char* mask      = (const unsigned char*)d_in[2];
    const float*         startT    = (const float*)d_in[3];
    const float*         endT      = (const float*)d_in[4];
    const float*         trans     = (const float*)d_in[5];
    float* out = (float*)d_out;
    float* num = (float*)d_ws;
    float* den = num + CRF_B;

    crf_numerator<<<CRF_B / 256, 256, 0, stream>>>(emissions, tags, mask, startT, endT, trans, num);
    crf_forward_wmma<<<CRF_B / (FWD_WAVES * 16), FWD_WAVES * 32, 0, stream>>>(
        emissions, mask, startT, endT, trans, den);
    crf_reduce<<<1, 256, 0, stream>>>(num, den, out);
}
